// PureLSTM_58265526337748
// MI455X (gfx1250) — matrix operand order repaired and checked by reading, not compiled, not run on hardware
//
#include <hip/hip_runtime.h>
#include <hip/hip_fp16.h>

typedef __attribute__((ext_vector_type(16))) _Float16 v16h;
typedef __attribute__((ext_vector_type(8)))  float    v8f;

#define WAVES 4
#define TSTEPS 10

union V16 { v16h v; unsigned u[8]; };

__device__ __forceinline__ float sigm_(float x) {
    return 1.0f / (1.0f + __expf(-x));
}
__device__ __forceinline__ float tanh_(float x) {
    float e = __expf(-2.0f * x);
    return (1.0f - e) / (1.0f + e);
}
__device__ __forceinline__ unsigned pack2_(float a, float b) {
    union { __half2 h2; unsigned u; } cv;
    cv.h2 = __floats2half2_rn(a, b);
    return cv.u;
}

// Rebuild the WMMA B-operand (h^T, 32xK16-layout) from lane-local gate values.
// Per lane: hv[t][v] holds h[j] for j = 16*t + v + 8*hi (hi = lane>>4).
// B-operand lane needs K = 16*hi + 2v'+hh  ->  exchange one half with lane^16.
__device__ __forceinline__ void build_B(V16& hB, const float hv[2][8], int hi) {
    unsigned t0p[4], t1p[4];
#pragma unroll
    for (int k = 0; k < 4; ++k) {
        t0p[k] = pack2_(hv[0][2 * k], hv[0][2 * k + 1]);
        t1p[k] = pack2_(hv[1][2 * k], hv[1][2 * k + 1]);
    }
#pragma unroll
    for (int k = 0; k < 4; ++k) {
        unsigned send = hi ? t0p[k] : t1p[k];
        unsigned recv = __shfl_xor(send, 16, 32);
        hB.u[k]     = hi ? recv   : t0p[k];
        hB.u[4 + k] = hi ? t1p[k] : recv;
    }
}

__global__ __launch_bounds__(32 * WAVES)
void lstm2_wmma_kernel(const float* __restrict__ x,
                       const float* __restrict__ W_ih0,
                       const float* __restrict__ W_hh0,
                       const float* __restrict__ b_ih0,
                       const float* __restrict__ b_hh0,
                       const float* __restrict__ W_ih1,
                       const float* __restrict__ W_hh1,
                       const float* __restrict__ b_ih1,
                       const float* __restrict__ b_hh1,
                       const float* __restrict__ W_fc,
                       const float* __restrict__ b_fc,
                       float* __restrict__ out,
                       int Btot) {
    // LDS: 3 weight matrices pre-swizzled into WMMA A-operand layout
    // [matrix(3)][tile(8)][lane(32)][16 halves]  = 24 KB
    __shared__ _Float16 wlds[3 * 8 * 32 * 16];
    __shared__ float bias0[128];  // b_ih0 + b_hh0
    __shared__ float wih0[128];   // W_ih0 column (I==1)
    __shared__ float bias1[128];  // b_ih1 + b_hh1
    __shared__ float wfc[32];

    const int tid = threadIdx.x;

    // ---- cooperative preload: swizzle W[g,k] (g=tile*16+(lane&15),
    //      k = 16*(lane>>4) + s) into A-layout ----
    for (int idx = tid; idx < 3 * 4096; idx += blockDim.x) {
        int m    = idx / 4096;
        int r    = idx & 4095;
        int tile = r >> 9;
        int lane = (r >> 4) & 31;
        int s    = r & 15;
        int v    = s >> 1, hh = s & 1;
        int hi   = lane >> 4;
        int g    = tile * 16 + (lane & 15);
        int k    = 16 * hi + s;
        const float* W = (m == 0) ? W_hh0 : (m == 1) ? W_ih1 : W_hh1;
        wlds[idx] = (_Float16)W[g * 32 + k];
    }
    for (int g = tid; g < 128; g += blockDim.x) {
        bias0[g] = b_ih0[g] + b_hh0[g];
        wih0[g]  = W_ih0[g];
        bias1[g] = b_ih1[g] + b_hh1[g];
    }
    if (tid < 32) wfc[tid] = W_fc[tid];
    __syncthreads();

    const int lane = tid & 31;
    const int wave = tid >> 5;
    const int hi   = lane >> 4;
    const long base = ((long)blockIdx.x * WAVES + wave) * 16;
    if (base >= Btot) return;              // wave-uniform exit (Btot % 16 == 0)
    const int row = (int)base + (lane & 15);

    // preload this lane's x sequence (I == 1)
    float xt[TSTEPS];
#pragma unroll
    for (int t = 0; t < TSTEPS; ++t) xt[t] = x[row * TSTEPS + t];

    // ================= layer 0 =================
    V16 hB;                                // h^T as WMMA B operand (f16)
#pragma unroll
    for (int i = 0; i < 8; ++i) hB.u[i] = 0;
    float c0[2][8];
#pragma unroll
    for (int p = 0; p < 2; ++p)
#pragma unroll
        for (int e = 0; e < 8; ++e) c0[p][e] = 0.0f;

    V16 h0s[TSTEPS];                       // stashed layer-0 outputs (B layout)

#pragma unroll
    for (int t = 0; t < TSTEPS; ++t) {
        float hv[2][8];
#pragma unroll
        for (int p = 0; p < 2; ++p) {
            v8f acc[4];
#pragma unroll
            for (int q = 0; q < 4; ++q) {          // i, f, g, o
                int tile = q * 2 + p;
                const float* bp = &bias0[tile * 16 + 8 * hi];
                const float* wp = &wih0[tile * 16 + 8 * hi];
                v8f ci;
#pragma unroll
                for (int e = 0; e < 8; ++e) ci[e] = bp[e] + xt[t] * wp[e];
                v16h A = *(const v16h*)&wlds[(0 * 8 + tile) * 512 + lane * 16];
                acc[q] = __builtin_amdgcn_wmma_f32_16x16x32_f16(
                    false, A, false, hB.v, (short)0, ci, false, false);
            }
#pragma unroll
            for (int e = 0; e < 8; ++e) {
                float ig = sigm_(acc[0][e]);
                float fg = sigm_(acc[1][e]);
                float gg = tanh_(acc[2][e]);
                float og = sigm_(acc[3][e]);
                float cc = fg * c0[p][e] + ig * gg;
                c0[p][e] = cc;
                hv[p][e] = og * tanh_(cc);
            }
        }
        build_B(hB, hv, hi);
        h0s[t] = hB;
    }

    // ================= layer 1 =================
    V16 h1B;
#pragma unroll
    for (int i = 0; i < 8; ++i) h1B.u[i] = 0;
    float c1[2][8];
#pragma unroll
    for (int p = 0; p < 2; ++p)
#pragma unroll
        for (int e = 0; e < 8; ++e) c1[p][e] = 0.0f;

    float hv1[2][8];
#pragma unroll
    for (int t = 0; t < TSTEPS; ++t) {
#pragma unroll
        for (int p = 0; p < 2; ++p) {
            v8f acc[4];
#pragma unroll
            for (int q = 0; q < 4; ++q) {
                int tile = q * 2 + p;
                const float* bp = &bias1[tile * 16 + 8 * hi];
                v8f ci;
#pragma unroll
                for (int e = 0; e < 8; ++e) ci[e] = bp[e];
                v16h Aih = *(const v16h*)&wlds[(1 * 8 + tile) * 512 + lane * 16];
                v8f tmp = __builtin_amdgcn_wmma_f32_16x16x32_f16(
                    false, Aih, false, h0s[t].v, (short)0, ci, false, false);
                v16h Ahh = *(const v16h*)&wlds[(2 * 8 + tile) * 512 + lane * 16];
                acc[q] = __builtin_amdgcn_wmma_f32_16x16x32_f16(
                    false, Ahh, false, h1B.v, (short)0, tmp, false, false);
            }
#pragma unroll
            for (int e = 0; e < 8; ++e) {
                float ig = sigm_(acc[0][e]);
                float fg = sigm_(acc[1][e]);
                float gg = tanh_(acc[2][e]);
                float og = sigm_(acc[3][e]);
                float cc = fg * c1[p][e] + ig * gg;
                c1[p][e] = cc;
                hv1[p][e] = og * tanh_(cc);
            }
        }
        build_B(h1B, hv1, hi);
    }

    // ================= FC + sigmoid =================
    float partial = 0.0f;
#pragma unroll
    for (int p = 0; p < 2; ++p) {
        const float* wp = &wfc[16 * p + 8 * hi];
#pragma unroll
        for (int e = 0; e < 8; ++e) partial += wp[e] * hv1[p][e];
    }
    float tot = partial + __shfl_xor(partial, 16, 32);
    tot = sigm_(tot + b_fc[0]);
    if (hi == 0) out[row] = tot;
}

extern "C" void kernel_launch(void* const* d_in, const int* in_sizes, int n_in,
                              void* d_out, int out_size, void* d_ws, size_t ws_size,
                              hipStream_t stream) {
    const float* x     = (const float*)d_in[0];
    const float* W_ih0 = (const float*)d_in[1];
    const float* W_hh0 = (const float*)d_in[2];
    const float* b_ih0 = (const float*)d_in[3];
    const float* b_hh0 = (const float*)d_in[4];
    const float* W_ih1 = (const float*)d_in[5];
    const float* W_hh1 = (const float*)d_in[6];
    const float* b_ih1 = (const float*)d_in[7];
    const float* b_hh1 = (const float*)d_in[8];
    const float* W_fc  = (const float*)d_in[9];
    const float* b_fc  = (const float*)d_in[10];
    float* out = (float*)d_out;

    int Btot   = in_sizes[0] / TSTEPS;          // x is [B, T, 1]
    int blocks = (Btot + (16 * WAVES - 1)) / (16 * WAVES);
    lstm2_wmma_kernel<<<blocks, 32 * WAVES, 0, stream>>>(
        x, W_ih0, W_hh0, b_ih0, b_hh0, W_ih1, W_hh1, b_ih1, b_hh1,
        W_fc, b_fc, out, Btot);
}
